// Block_mamba_16063177687681
// MI455X (gfx1250) — matrix operand order repaired and checked by reading, not compiled, not run on hardware
//
#include <hip/hip_runtime.h>
#include <hip/hip_bf16.h>

// ---------------------------------------------------------------------------
// MI455X (gfx1250) implementation of the Mamba+EinFFT block.
//
// Precision strategy: activations/weights in bf16 for v_wmma_f32_16x16x32_bf16
// (f32 accumulate). The 1024-pt FFT is realized as a DFT-matrix GEMM so all
// heavy math rides the WMMA pipe. Elementwise ops (LN, conv+SiLU, scan,
// radix-4 FFT stage, gates) stay in f32 on the VALU.
// ---------------------------------------------------------------------------

typedef __bf16 bf16_t;
typedef __bf16 bf16x16 __attribute__((ext_vector_type(16)));
typedef __bf16 bf16x4  __attribute__((ext_vector_type(4)));
typedef float  v8f     __attribute__((ext_vector_type(8)));
typedef unsigned int u32x4 __attribute__((ext_vector_type(4)));

#define DEV __device__ __forceinline__

DEV v8f wmma_bf16(bf16x16 a, bf16x16 b, v8f c) {
  // D = A(16x32) * B(32x16) + C, f32 accumulate
  return __builtin_amdgcn_wmma_f32_16x16x32_bf16(false, a, false, b, (short)0, c,
                                                 false, false);
}

// A fragment (row-major bf16, 16x32 tile): lane l holds row (l&15);
// elems 0..7 = K [h*8, h*8+8), elems 8..15 = K [16+h*8, 16+h*8+8), h = l>>4.
DEV bf16x16 load_frag_a(const bf16_t* __restrict__ A, int lda, int row, int h, int k0) {
  const bf16_t* p = A + (size_t)row * lda + k0 + h * 8;
  union { u32x4 q[2]; bf16x16 v; } u;
  u.q[0] = *(const u32x4*)p;
  u.q[1] = *(const u32x4*)(p + 16);
  return u.v;
}
// B fragment (weights stored (ncol, K) row-major): lane l holds col (l&15),
// elems 0..7 = K [h*8, h*8+8), elems 8..15 = K [16+h*8, 16+h*8+8), h = l>>4.
DEV bf16x16 load_frag_b(const bf16_t* __restrict__ B, int ldb, int col, int h, int k0) {
  const bf16_t* p = B + (size_t)col * ldb + k0 + h * 8;
  union { u32x4 q[2]; bf16x16 v; } u;
  u.q[0] = *(const u32x4*)p;
  u.q[1] = *(const u32x4*)(p + 16);
  return u.v;
}

enum { EPI_NONE = 0, EPI_RESID = 1, EPI_SOFTPLUS = 2, EPI_RELU = 3, EPI_SHRINK_T = 4 };

// Generic dual-source WMMA GEMM:
//   acc = A1 @ B1^T  (+ sign2 * A2 @ B2^T  when DUAL)
// A row-major bf16 (M x K), B row-major bf16 (N x K). Block tile 64x64,
// 4 waves (2x2), each wave 2x2 WMMA tiles. blockIdx.z selects a slice via
// element strides (batch / frequency-block fusion). DUAL is a template param
// so the hot K-loop is branch-free.
template <int EPI, bool DUAL>
__global__ __launch_bounds__(128) void gemm_wmma(
    const bf16_t* __restrict__ A1, const bf16_t* __restrict__ B1,
    const bf16_t* __restrict__ A2, const bf16_t* __restrict__ B2, float sign2,
    int lda, int ldb, int K,
    float* __restrict__ outF, bf16_t* __restrict__ outB, int ldo,
    const float* __restrict__ bias, const float* __restrict__ resid, int ldr,
    float lam,
    long long zsA, long long zsB, long long zsO, long long zsBias) {
  const long long z = (long long)blockIdx.z;
  A1 += z * zsA; B1 += z * zsB;
  if (DUAL) { A2 += z * zsA; B2 += z * zsB; }
  if (outF) outF += z * zsO;
  if (outB) outB += z * zsO;
  if (bias) bias += z * zsBias;

  const int wave = threadIdx.x >> 5, lane = threadIdx.x & 31;
  const int wm = wave >> 1, wn = wave & 1;
  const int row0 = blockIdx.x * 64 + wm * 32;
  const int col0 = blockIdx.y * 64 + wn * 32;
  const int t16 = lane & 15, h = lane >> 4;

  v8f acc[2][2], acc2[2][2];
  for (int i = 0; i < 2; i++)
    for (int j = 0; j < 2; j++)
      for (int e = 0; e < 8; e++) { acc[i][j][e] = 0.f; acc2[i][j][e] = 0.f; }

  for (int k0 = 0; k0 < K; k0 += 32) {
    bf16x16 a0 = load_frag_a(A1, lda, row0 + t16,      h, k0);
    bf16x16 a1 = load_frag_a(A1, lda, row0 + 16 + t16, h, k0);
    bf16x16 b0 = load_frag_b(B1, ldb, col0 + t16,      h, k0);
    bf16x16 b1 = load_frag_b(B1, ldb, col0 + 16 + t16, h, k0);
    acc[0][0] = wmma_bf16(a0, b0, acc[0][0]);
    acc[0][1] = wmma_bf16(a0, b1, acc[0][1]);
    acc[1][0] = wmma_bf16(a1, b0, acc[1][0]);
    acc[1][1] = wmma_bf16(a1, b1, acc[1][1]);
    if (DUAL) {
      bf16x16 c0 = load_frag_a(A2, lda, row0 + t16,      h, k0);
      bf16x16 c1 = load_frag_a(A2, lda, row0 + 16 + t16, h, k0);
      bf16x16 d0 = load_frag_b(B2, ldb, col0 + t16,      h, k0);
      bf16x16 d1 = load_frag_b(B2, ldb, col0 + 16 + t16, h, k0);
      acc2[0][0] = wmma_bf16(c0, d0, acc2[0][0]);
      acc2[0][1] = wmma_bf16(c0, d1, acc2[0][1]);
      acc2[1][0] = wmma_bf16(c1, d0, acc2[1][0]);
      acc2[1][1] = wmma_bf16(c1, d1, acc2[1][1]);
    }
  }

  // D layout: lane l, VGPR r -> element (m = h*8 + r, n = l&15) of 16x16 tile.
  for (int ti = 0; ti < 2; ti++) {
    for (int tj = 0; tj < 2; tj++) {
      const int orow = row0 + ti * 16 + h * 8;
      const int ocol = col0 + tj * 16 + t16;
#pragma unroll
      for (int r = 0; r < 8; r++) {
        float v = acc[ti][tj][r];
        if (DUAL) v += sign2 * acc2[ti][tj][r];
        if (EPI == EPI_RESID)    v += resid[(size_t)(orow + r) * ldr + ocol];
        if (EPI == EPI_SOFTPLUS) { v += bias[ocol]; v = (v > 20.f) ? v : log1pf(__expf(v)); }
        if (EPI == EPI_RELU)     { v += bias[ocol]; v = fmaxf(v, 0.f); }
        if (EPI == EPI_SHRINK_T) {
          v += bias[ocol];
          v = (v > lam) ? (v - lam) : ((v < -lam) ? (v + lam) : 0.f);
          // transposed store is CONTIGUOUS per lane in the D layout
          outB[(size_t)ocol * ldo + orow + r] = (bf16_t)v;
        } else {
          if (outF) outF[(size_t)(orow + r) * ldo + ocol] = v;
          if (outB) outB[(size_t)(orow + r) * ldo + ocol] = (bf16_t)v;
        }
      }
    }
  }
}

// ------------------------- elementwise / helper kernels ---------------------

// LayerNorm over C=512, one 128-thread block per token (4 elems/thread).
__global__ __launch_bounds__(128) void layernorm_kernel(
    const float* __restrict__ x, const float* __restrict__ w,
    const float* __restrict__ b, float* __restrict__ outF,
    bf16_t* __restrict__ outB) {
  const int tok = blockIdx.x, tid = threadIdx.x;
  const float4 v = *(const float4*)(x + (size_t)tok * 512 + tid * 4);
  float s1 = v.x + v.y + v.z + v.w;
  float s2 = v.x * v.x + v.y * v.y + v.z * v.z + v.w * v.w;
  for (int off = 16; off; off >>= 1) {
    s1 += __shfl_xor(s1, off, 32);
    s2 += __shfl_xor(s2, off, 32);
  }
  __shared__ float sm[8];
  const int wave = tid >> 5, lane = tid & 31;
  if (lane == 0) { sm[wave] = s1; sm[4 + wave] = s2; }
  __syncthreads();
  const float t1 = sm[0] + sm[1] + sm[2] + sm[3];
  const float t2 = sm[4] + sm[5] + sm[6] + sm[7];
  const float mean = t1 * (1.f / 512.f);
  const float var = t2 * (1.f / 512.f) - mean * mean;
  const float rstd = rsqrtf(var + 1e-5f);
  const int c = tid * 4;
  const float4 wv = *(const float4*)(w + c);
  const float4 bv = *(const float4*)(b + c);
  float o0 = (v.x - mean) * rstd * wv.x + bv.x;
  float o1 = (v.y - mean) * rstd * wv.y + bv.y;
  float o2 = (v.z - mean) * rstd * wv.z + bv.z;
  float o3 = (v.w - mean) * rstd * wv.w + bv.w;
  float4 of = {o0, o1, o2, o3};
  *(float4*)(outF + (size_t)tok * 512 + c) = of;
  bf16x4 ob; ob[0] = (bf16_t)o0; ob[1] = (bf16_t)o1; ob[2] = (bf16_t)o2; ob[3] = (bf16_t)o3;
  *(bf16x4*)(outB + (size_t)tok * 512 + c) = ob;
}

// Depthwise causal conv (DC=4) over xm = xz[:, :1024], + bias, + SiLU.
__global__ __launch_bounds__(256) void conv_silu_kernel(
    const float* __restrict__ xz, const float* __restrict__ cw,
    const float* __restrict__ cb, float* __restrict__ xactF,
    bf16_t* __restrict__ xactB) {
  const int idx = blockIdx.x * 256 + threadIdx.x;  // 2*1024*1024
  const int d = idx & 1023, t = (idx >> 10) & 1023, b = idx >> 20;
  float acc = cb[d];
#pragma unroll
  for (int j = 0; j < 4; j++) {
    const int tt = t - 3 + j;
    if (tt >= 0) acc += cw[d * 4 + j] * xz[((size_t)b * 1024 + tt) * 2048 + d];
  }
  const float s = acc / (1.f + __expf(-acc));  // SiLU
  xactF[(size_t)idx] = s;
  xactB[(size_t)idx] = (bf16_t)s;
}

// Selective scan: one wave32 per (b, channel d), 2 states per lane (DS=64).
// Fuses the D skip connection and the SiLU(z) gate; emits bf16 for out_proj.
__global__ __launch_bounds__(256) void scan_kernel(
    const float* __restrict__ dt, const float* __restrict__ proj,
    const float* __restrict__ xact, const float* __restrict__ xz,
    const float* __restrict__ A_log, const float* __restrict__ Dv,
    bf16_t* __restrict__ ygB) {
  const int wave = threadIdx.x >> 5, lane = threadIdx.x & 31;
  const int gi = blockIdx.x * 8 + wave;  // 2048 (b,d) channels
  const int b = gi >> 10, d = gi & 1023;
  const int s0 = lane, s1 = lane + 32;
  const float A0 = -__expf(A_log[d * 64 + s0]);
  const float A1 = -__expf(A_log[d * 64 + s1]);
  const float Dd = Dv[d];
  float h0 = 0.f, h1 = 0.f;
  for (int t = 0; t < 1024; t++) {
    const size_t tok = (size_t)b * 1024 + t;
    const float dtv = dt[tok * 1024 + d];
    const float u = xact[tok * 1024 + d];
    const float Bm0 = proj[tok * 192 + 32 + s0], Bm1 = proj[tok * 192 + 32 + s1];
    const float Cm0 = proj[tok * 192 + 96 + s0], Cm1 = proj[tok * 192 + 96 + s1];
    h0 = __expf(dtv * A0) * h0 + dtv * Bm0 * u;
    h1 = __expf(dtv * A1) * h1 + dtv * Bm1 * u;
    float y = h0 * Cm0 + h1 * Cm1;
    for (int off = 16; off; off >>= 1) y += __shfl_xor(y, off, 32);
    if (lane == 0) {
      const float zz = xz[tok * 2048 + 1024 + d];
      const float g = zz / (1.f + __expf(-zz));  // SiLU gate
      ygB[tok * 1024 + d] = (bf16_t)((y + u * Dd) * g);
    }
  }
}

// Forward radix-4 FFT along NB (ortho 1/2) + transpose to channel-major bf16.
__global__ __launch_bounds__(256) void fft4_fwd_kernel(
    const float* __restrict__ xn, bf16_t* __restrict__ Xr,
    bf16_t* __restrict__ Xi) {
  const int idx = blockIdx.x * 256 + threadIdx.x;  // 2*128*1024
  const int n = idx & 1023, d = (idx >> 10) & 127, b = idx >> 17;
  const float* p = xn + ((size_t)b * 1024 + n) * 512 + d;
  const float x0 = p[0], x1 = p[128], x2 = p[256], x3 = p[384];
  const float e = x0 + x2, o = x0 - x2, e2 = x1 + x3, o2 = x1 - x3;
  const size_t base = ((size_t)b * 512 + d) * 1024 + n;  // channel-major
  Xr[base]              = (bf16_t)(0.5f * (e + e2));
  Xi[base]              = (bf16_t)0.f;
  Xr[base + 131072]     = (bf16_t)(0.5f * o);
  Xi[base + 131072]     = (bf16_t)(-0.5f * o2);
  Xr[base + 2 * 131072] = (bf16_t)(0.5f * (e - e2));
  Xi[base + 2 * 131072] = (bf16_t)0.f;
  Xr[base + 3 * 131072] = (bf16_t)(0.5f * o);
  Xi[base + 3 * 131072] = (bf16_t)(0.5f * o2);
}

// Inverse radix-4 FFT (ortho 1/2), real part, + residual add.
__global__ __launch_bounds__(256) void ifft4_real_add_kernel(
    const float* __restrict__ Zr, const float* __restrict__ Zi,
    float* __restrict__ xcur, float* __restrict__ out2) {
  const int idx = blockIdx.x * 256 + threadIdx.x;  // 2*1024*128
  const int d = idx & 127, n = (idx >> 7) & 1023, b = idx >> 17;
  const size_t base = ((size_t)b * 1024 + n) * 512 + d;
  const float r0 = Zr[base], r1 = Zr[base + 128], r2 = Zr[base + 256], r3 = Zr[base + 384];
  const float i1 = Zi[base + 128], i3 = Zi[base + 384];
  const float y0 = 0.5f * (r0 + r1 + r2 + r3);
  const float y1 = 0.5f * (r0 - i1 - r2 + i3);
  const float y2 = 0.5f * (r0 - r1 + r2 - r3);
  const float y3 = 0.5f * (r0 + i1 - r2 - i3);
  const float o0 = xcur[base] + y0;
  const float o1 = xcur[base + 128] + y1;
  const float o2 = xcur[base + 256] + y2;
  const float o3 = xcur[base + 384] + y3;
  xcur[base] = o0; xcur[base + 128] = o1; xcur[base + 256] = o2; xcur[base + 384] = o3;
  if (out2) { out2[base] = o0; out2[base + 128] = o1; out2[base + 256] = o2; out2[base + 384] = o3; }
}

// ------------------------------ prep kernels --------------------------------

__global__ void cvt_bf16_kernel(const float* __restrict__ s, bf16_t* __restrict__ d, int n) {
  const int i = blockIdx.x * 256 + threadIdx.x;
  if (i < n) d[i] = (bf16_t)s[i];
}

// x_proj_w (160,1024) -> zero-padded (192,1024) bf16
__global__ void pad_xproj_kernel(const float* __restrict__ s, bf16_t* __restrict__ d) {
  const int i = blockIdx.x * 256 + threadIdx.x;  // 192*1024
  const int row = i >> 10;
  d[i] = (row < 160) ? (bf16_t)s[i] : (bf16_t)0.f;
}

// cw (2,4,128,128)[i][nb][d][k] -> (2,4,128,128)[i][nb][k][d] bf16
__global__ void cwt_kernel(const float* __restrict__ s, bf16_t* __restrict__ d) {
  const int i = blockIdx.x * 256 + threadIdx.x;  // 131072
  const int dd = i & 127, k = (i >> 7) & 127, nb = (i >> 14) & 3, ri = i >> 16;
  d[i] = (bf16_t)s[(((size_t)(ri * 4 + nb) * 128 + dd) * 128) + k];
}

// DFT matrix, ortho 1/32: Fr = cos(2*pi*k*n/1024)/32, Fi = -sin(...)/32.
__global__ void fgen_kernel(bf16_t* __restrict__ Fr, bf16_t* __restrict__ Fi) {
  const int idx = blockIdx.x * 256 + threadIdx.x;  // 1024*1024
  const int n = idx & 1023, k = idx >> 10;
  const int p = (k * n) & 1023;
  const float ang = (float)p * (6.283185307179586f / 1024.f);
  Fr[idx] = (bf16_t)(__cosf(ang) * (1.f / 32.f));
  Fi[idx] = (bf16_t)(-__sinf(ang) * (1.f / 32.f));
}

// ------------------------------- host driver --------------------------------

extern "C" void kernel_launch(void* const* d_in, const int* in_sizes, int n_in,
                              void* d_out, int out_size, void* d_ws, size_t ws_size,
                              hipStream_t stream) {
  (void)in_sizes; (void)n_in; (void)out_size;
  const float* x_in = (const float*)d_in[0];
  const float* ln_w = (const float*)d_in[1];
  const float* ln_b = (const float*)d_in[2];
  const float* in_proj_w = (const float*)d_in[3];
  const float* conv_w = (const float*)d_in[4];
  const float* conv_b = (const float*)d_in[5];
  const float* x_proj_w = (const float*)d_in[6];
  const float* dt_proj_w = (const float*)d_in[7];
  const float* dt_proj_b = (const float*)d_in[8];
  const float* A_log = (const float*)d_in[9];
  const float* Dvec = (const float*)d_in[10];
  const float* out_proj_w = (const float*)d_in[11];
  const float* n2_w = (const float*)d_in[12];
  const float* n2_b = (const float*)d_in[13];
  const float* cw1 = (const float*)d_in[14];
  const float* cw2 = (const float*)d_in[15];
  const float* cb1 = (const float*)d_in[16];
  const float* cb2 = (const float*)d_in[17];

  char* ws = (char*)d_ws;
  size_t off = 0;
  auto alloc = [&](size_t bytes) -> char* {
    char* p = ws + off;
    off = (off + bytes + 255) & ~(size_t)255;
    return p;
  };
  // bf16 weight copies
  bf16_t* WinB = (bf16_t*)alloc(2048 * 512 * 2);
  bf16_t* WxpB = (bf16_t*)alloc(192 * 1024 * 2);
  bf16_t* WdtB = (bf16_t*)alloc(1024 * 32 * 2);
  bf16_t* WouB = (bf16_t*)alloc(512 * 1024 * 2);
  bf16_t* cw1t = (bf16_t*)alloc(131072 * 2);
  bf16_t* cw2t = (bf16_t*)alloc(131072 * 2);
  bf16_t* FrB = (bf16_t*)alloc(1024 * 1024 * 2);
  bf16_t* FiB = (bf16_t*)alloc(1024 * 1024 * 2);
  // persistent activations
  float* xcur = (float*)alloc(1048576 * 4);
  float* xnF = (float*)alloc(1048576 * 4);
  bf16_t* xnB = (bf16_t*)alloc(1048576 * 2);
  float* projF = (float*)alloc(393216 * 4);
  bf16_t* projB = (bf16_t*)alloc(393216 * 2);
  bf16_t* xactB = (bf16_t*)alloc(2097152 * 2);
  bf16_t* ygB = (bf16_t*)alloc(2097152 * 2);
  // 32MB pool shared between mamba temps and einfft temps (disjoint lifetimes)
  char* pool = alloc(32u * 1024 * 1024);
  float* xz = (float*)pool;                          // 16MB (mamba)
  float* xactF = (float*)(pool + 16777216);          //  8MB (mamba)
  float* dtF = (float*)(pool + 25165824);            //  8MB (mamba)
  bf16_t* XRC = (bf16_t*)(pool + 0);                 // einfft view
  bf16_t* XIC = (bf16_t*)(pool + 2097152);
  bf16_t* YRB = (bf16_t*)(pool + 4194304);
  bf16_t* YIB = (bf16_t*)(pool + 6291456);
  bf16_t* R1B = (bf16_t*)(pool + 8388608);
  bf16_t* I1B = (bf16_t*)(pool + 10485760);
  bf16_t* R2T = (bf16_t*)(pool + 12582912);
  bf16_t* I2T = (bf16_t*)(pool + 14680064);
  float* ZR = (float*)(pool + 16777216);
  float* ZI = (float*)(pool + 20971520);

  if (off > ws_size) return;  // workspace too small: no-op (compile-only loop)

  // ---- prep: bf16 weight conversions, cw transpose, DFT matrix ----
  cvt_bf16_kernel<<<4096, 256, 0, stream>>>(in_proj_w, WinB, 1048576);
  pad_xproj_kernel<<<768, 256, 0, stream>>>(x_proj_w, WxpB);
  cvt_bf16_kernel<<<128, 256, 0, stream>>>(dt_proj_w, WdtB, 32768);
  cvt_bf16_kernel<<<2048, 256, 0, stream>>>(out_proj_w, WouB, 524288);
  cwt_kernel<<<512, 256, 0, stream>>>(cw1, cw1t);
  cwt_kernel<<<512, 256, 0, stream>>>(cw2, cw2t);
  fgen_kernel<<<4096, 256, 0, stream>>>(FrB, FiB);
  hipMemcpyAsync(xcur, x_in, 1048576 * 4, hipMemcpyDeviceToDevice, stream);

  const long long ZB = 524288;  // per-batch element stride (1024*512)
  for (int blk = 0; blk < 2; blk++) {
    // ---------------- Mamba ----------------
    layernorm_kernel<<<2048, 128, 0, stream>>>(xcur, ln_w, ln_b, xnF, xnB);
    // xz = xn @ in_proj_w^T   (2048 x 512 x 2048)
    gemm_wmma<EPI_NONE, false><<<dim3(32, 32, 1), 128, 0, stream>>>(
        xnB, WinB, nullptr, nullptr, 0.f, 512, 512, 512, xz, nullptr, 2048,
        nullptr, nullptr, 0, 0.f, 0, 0, 0, 0);
    conv_silu_kernel<<<8192, 256, 0, stream>>>(xz, conv_w, conv_b, xactF, xactB);
    // proj = xact @ x_proj_w^T (2048 x 1024 x 192pad)
    gemm_wmma<EPI_NONE, false><<<dim3(32, 3, 1), 128, 0, stream>>>(
        xactB, WxpB, nullptr, nullptr, 0.f, 1024, 1024, 1024, projF, projB, 192,
        nullptr, nullptr, 0, 0.f, 0, 0, 0, 0);
    // dt = softplus(proj[:, :32] @ dt_proj_w^T + b)  (2048 x 32 x 1024)
    gemm_wmma<EPI_SOFTPLUS, false><<<dim3(32, 16, 1), 128, 0, stream>>>(
        projB, WdtB, nullptr, nullptr, 0.f, 192, 32, 32, dtF, nullptr, 1024,
        dt_proj_b, nullptr, 0, 0.f, 0, 0, 0, 0);
    scan_kernel<<<256, 256, 0, stream>>>(dtF, projF, xactF, xz, A_log, Dvec, ygB);
    // xcur += yg @ out_proj_w^T (2048 x 1024 x 512), residual fused
    gemm_wmma<EPI_RESID, false><<<dim3(32, 8, 1), 128, 0, stream>>>(
        ygB, WouB, nullptr, nullptr, 0.f, 1024, 1024, 1024, xcur, nullptr, 512,
        nullptr, xcur, 512, 0.f, 0, 0, 0, 0);

    // ---------------- EinFFT ----------------
    layernorm_kernel<<<2048, 128, 0, stream>>>(xcur, n2_w, n2_b, xnF, xnB);
    fft4_fwd_kernel<<<1024, 256, 0, stream>>>(xnF, XRC, XIC);
    // forward 1024-pt DFT as GEMM (per batch via grid.z)
    gemm_wmma<EPI_NONE, true><<<dim3(16, 8, 2), 128, 0, stream>>>(
        FrB, XRC, FiB, XIC, -1.f, 1024, 1024, 1024, nullptr, YRB, 512,
        nullptr, nullptr, 0, 0.f, 0, ZB, ZB, 0);
    gemm_wmma<EPI_NONE, true><<<dim3(16, 8, 2), 128, 0, stream>>>(
        FrB, XIC, FiB, XRC, 1.f, 1024, 1024, 1024, nullptr, YIB, 512,
        nullptr, nullptr, 0, 0.f, 0, ZB, ZB, 0);
    // complex mixing 1: relu(Yr@w0 - Yi@w1 + b0), relu(Yr@w1 + Yi@w0 + b1)
    gemm_wmma<EPI_RELU, true><<<dim3(32, 2, 4), 128, 0, stream>>>(
        YRB, cw1t, YIB, cw1t + 65536, -1.f, 512, 128, 128, nullptr, R1B, 512,
        cb1, nullptr, 0, 0.f, 128, 16384, 128, 128);
    gemm_wmma<EPI_RELU, true><<<dim3(32, 2, 4), 128, 0, stream>>>(
        YRB, cw1t + 65536, YIB, cw1t, 1.f, 512, 128, 128, nullptr, I1B, 512,
        cb1 + 512, nullptr, 0, 0.f, 128, 16384, 128, 128);
    // complex mixing 2 + softshrink, stored TRANSPOSED (channel-major) bf16
    gemm_wmma<EPI_SHRINK_T, true><<<dim3(32, 2, 4), 128, 0, stream>>>(
        R1B, cw2t, I1B, cw2t + 65536, -1.f, 512, 128, 128, nullptr, R2T, 2048,
        cb2, nullptr, 0, 0.01f, 128, 16384, 262144, 128);
    gemm_wmma<EPI_SHRINK_T, true><<<dim3(32, 2, 4), 128, 0, stream>>>(
        R1B, cw2t + 65536, I1B, cw2t, 1.f, 512, 128, 128, nullptr, I2T, 2048,
        cb2 + 512, nullptr, 0, 0.01f, 128, 16384, 262144, 128);
    // inverse DFT: Zr = Fr@r2 + Fi@i2 ; Zi = Fr@i2 - Fi@r2
    gemm_wmma<EPI_NONE, true><<<dim3(16, 8, 2), 128, 0, stream>>>(
        FrB, R2T, FiB, I2T, 1.f, 1024, 2048, 1024, ZR, nullptr, 512,
        nullptr, nullptr, 0, 0.f, 0, 1024, ZB, 0);
    gemm_wmma<EPI_NONE, true><<<dim3(16, 8, 2), 128, 0, stream>>>(
        FrB, I2T, FiB, R2T, -1.f, 1024, 2048, 1024, ZI, nullptr, 512,
        nullptr, nullptr, 0, 0.f, 0, 1024, ZB, 0);
    ifft4_real_add_kernel<<<1024, 256, 0, stream>>>(
        ZR, ZI, xcur, (blk == 1) ? (float*)d_out : nullptr);
  }
}
